// Net_25984552141404
// MI455X (gfx1250) — matrix operand order repaired and checked by reading, not compiled, not run on hardware
//
#include <hip/hip_runtime.h>

typedef __bf16 bf16_t;
typedef __attribute__((ext_vector_type(16))) __bf16 v16bf;
typedef __attribute__((ext_vector_type(8)))  float  v8f;

// Problem constants
constexpr int kS  = 128;
constexpr int kB  = 8192;
constexpr int kE  = 300;
constexpr int kPADTOK = 1;
constexpr int kH1 = 2048;
constexpr int kH2 = 2048;
constexpr int kC  = 1221;
constexpr int kK1 = 608;    // 2*E = 600 padded to multiple of 32
constexpr int kNC = 1280;   // C padded to multiple of 128

// LDS tile: 128 rows x 32 bf16, row stride 40 elements (16B pad -> conflict-light b128 reads)
constexpr int kLdsStride = 40;

union Frag16 { v16bf v; uint4 q[2]; };

__device__ __forceinline__ bf16_t to_bf16(float f) { return (bf16_t)f; }

// --- CDNA5 async global->LDS copy (ASYNCcnt-tracked, direct-to-LDS path) ----
__device__ __forceinline__ void async_g2l_b128(bf16_t* lds_ptr, const bf16_t* gaddr) {
    unsigned lds_off = (unsigned)(size_t)lds_ptr;   // generic LDS addr: offset in [31:0]
    asm volatile("global_load_async_to_lds_b128 %0, %1, off"
                 :: "v"(lds_off), "v"(gaddr) : "memory");
}
__device__ __forceinline__ void wait_async4() {
    asm volatile("s_wait_asynccnt 0x4" ::: "memory");
}
__device__ __forceinline__ void wait_async0() {
    asm volatile("s_wait_asynccnt 0x0" ::: "memory");
}
__device__ __forceinline__ void wait_ds0() {
    asm volatile("s_wait_dscnt 0x0" ::: "memory");
}

// ---------------------------------------------------------------------------
// fp32 -> bf16 weight conversion with zero padding.
// ---------------------------------------------------------------------------
__global__ void convert_pad_kernel(const float* __restrict__ W, bf16_t* __restrict__ Wb,
                                   int rows, int cols, int prows, int pcols) {
    int idx = blockIdx.x * 256 + threadIdx.x;
    int total = prows * pcols;
    if (idx >= total) return;
    int r = idx / pcols;
    int c = idx - r * pcols;
    float v = (r < rows && c < cols) ? W[(size_t)r * cols + c] : 0.0f;
    Wb[idx] = to_bf16(v);
}

// ---------------------------------------------------------------------------
// Embedding + masked mean -> feat [B, kK1] bf16. One wave32 per sentence.
// ---------------------------------------------------------------------------
__global__ __launch_bounds__(256) void embed_feat_kernel(const int* __restrict__ x,
                                                         const float* __restrict__ emb,
                                                         bf16_t* __restrict__ feat) {
    int b    = blockIdx.x * 8 + (threadIdx.x >> 5);
    int lane = threadIdx.x & 31;

    int lmax = -1;
    for (int s = lane; s < kS; s += 32) {
        int tok = x[s * kB + b];
        if (tok != kPADTOK) lmax = s;
    }
    #pragma unroll
    for (int off = 16; off; off >>= 1) {
        int o = __shfl_xor(lmax, off, 32);
        lmax = lmax > o ? lmax : o;
    }
    if (lmax < 0) lmax = kS - 1;        // all-PAD sentence: reference keeps full length
    float inv_cnt = 1.0f / (float)(lmax + 1);

    float acc[10];
    #pragma unroll
    for (int i = 0; i < 10; ++i) acc[i] = 0.0f;

    for (int s = 0; s <= lmax; ++s) {
        int tok = x[s * kB + b];
        const float* row = emb + (size_t)tok * kE;
        #pragma unroll
        for (int i = 0; i < 10; ++i) {
            int e = lane + i * 32;
            if (e < kE) acc[i] += row[e];
        }
    }

    int tok0 = x[b];
    const float* row0 = emb + (size_t)tok0 * kE;
    bf16_t* outr = feat + (size_t)b * kK1;
    #pragma unroll
    for (int i = 0; i < 10; ++i) {
        int e = lane + i * 32;
        if (e < kE) {
            outr[e]       = to_bf16(row0[e]);
            outr[kE + e]  = to_bf16(acc[i] * inv_cnt);
        }
    }
    if (lane < kK1 - 2 * kE) outr[2 * kE + lane] = to_bf16(0.0f);
}

// ---------------------------------------------------------------------------
// Tiled WMMA GEMM, double-buffered async global->LDS staging.
// Block tile 128x128, 8 wave32 waves, each wave 32x64 via 2x4 wmma tiles.
// ---------------------------------------------------------------------------
__device__ __forceinline__ void issue_tile(const bf16_t* __restrict__ A,
                                           const bf16_t* __restrict__ W,
                                           int K, int bm, int bn, int k0,
                                           bf16_t* as, bf16_t* bs, int t) {
    #pragma unroll
    for (int it = 0; it < 2; ++it) {
        int i   = t + it * 256;         // 0..511
        int row = i >> 2;
        int seg = i & 3;
        async_g2l_b128(as + row * kLdsStride + seg * 8,
                       A + (size_t)(bm + row) * K + k0 + seg * 8);
        async_g2l_b128(bs + row * kLdsStride + seg * 8,
                       W + (size_t)(bn + row) * K + k0 + seg * 8);
    }
}

__device__ __forceinline__ void load_frags(const bf16_t* as, const bf16_t* bs,
                                           int wm, int wn, int arow, int abase,
                                           int bcol, int bbase,
                                           Frag16 (&af)[2], Frag16 (&bfr)[4]) {
    #pragma unroll
    for (int tm = 0; tm < 2; ++tm) {
        const bf16_t* p = as + (wm + tm * 16 + arow) * kLdsStride + abase;
        af[tm].q[0] = *(const uint4*)(p);
        af[tm].q[1] = *(const uint4*)(p + 16);   // K: base+16 .. base+23
    }
    #pragma unroll
    for (int tn = 0; tn < 4; ++tn) {
        const bf16_t* p = bs + (wn + tn * 16 + bcol) * kLdsStride + bbase;
        bfr[tn].q[0] = *(const uint4*)(p);
        bfr[tn].q[1] = *(const uint4*)(p + 16);
    }
}

__device__ __forceinline__ void do_wmma(const Frag16 (&af)[2], const Frag16 (&bfr)[4],
                                        v8f (&acc)[2][4]) {
    #pragma unroll
    for (int tn = 0; tn < 4; ++tn)
        #pragma unroll
        for (int tm = 0; tm < 2; ++tm)
            acc[tm][tn] = __builtin_amdgcn_wmma_f32_16x16x32_bf16(
                false, af[tm].v, false, bfr[tn].v,
                (short)0, acc[tm][tn], false, false);
}

template <bool RELU_BF16>
__global__ __launch_bounds__(256) void gemm_wmma_kernel(
        const bf16_t* __restrict__ A, const bf16_t* __restrict__ W,
        const float* __restrict__ bias,
        bf16_t* __restrict__ outb, float* __restrict__ outf,
        int Npad, int Nout, int K) {
    __shared__ bf16_t As[2][128 * kLdsStride];
    __shared__ bf16_t Bs[2][128 * kLdsStride];

    const int t    = threadIdx.x;
    const int lane = t & 31;
    const int wid  = t >> 5;
    const int bm   = blockIdx.y * 128;
    const int bn   = blockIdx.x * 128;
    const int wm   = (wid >> 1) * 32;   // 4 wave-rows  * 32
    const int wn   = (wid & 1) * 64;    // 2 wave-cols  * 64

    v8f acc[2][4] = {};

    const int arow  = lane & 15;
    const int abase = (lane >> 4) * 8;   // A frag K offset (ISA 16-bit A layout)
    const int bcol  = lane & 15;
    const int bbase = (lane >> 4) * 8;   // B frag K offset (ISA 16-bit B layout)

    const int nT = K >> 5;               // K-steps of 32 (>= 2 for all layers)

    issue_tile(A, W, K, bm, bn, 0, As[0], Bs[0], t);     // preload tile 0

    Frag16 af[2], bfr[4];

    // Steady state: branch-free. While tile i is computed, tile i+1 is in flight.
    for (int i = 0; i < nT - 1; ++i) {
        issue_tile(A, W, K, bm, bn, (i + 1) * 32, As[(i + 1) & 1], Bs[(i + 1) & 1], t);
        wait_async4();                   // tile i landed (in-order); tile i+1 in flight
        __syncthreads();                 // all waves' shares of tile i visible

        load_frags(As[i & 1], Bs[i & 1], wm, wn, arow, abase, bcol, bbase, af, bfr);
        wait_ds0();                      // fragments in VGPRs
        __syncthreads();                 // buf[i&1] free for overwrite next iteration

        do_wmma(af, bfr, acc);           // matrix pipe runs while loop issues next tile
    }

    // Peeled last tile
    wait_async0();
    __syncthreads();
    load_frags(As[(nT - 1) & 1], Bs[(nT - 1) & 1], wm, wn, arow, abase, bcol, bbase, af, bfr);
    do_wmma(af, bfr, acc);

    // --- epilogue: bias (+ ReLU -> bf16) or fp32 store with N guard ---
    #pragma unroll
    for (int tm = 0; tm < 2; ++tm) {
        #pragma unroll
        for (int tn = 0; tn < 4; ++tn) {
            int n = bn + wn + tn * 16 + (lane & 15);
            float bv = (n < Nout) ? bias[n] : 0.0f;
            #pragma unroll
            for (int r = 0; r < 8; ++r) {
                int m = bm + wm + tm * 16 + (lane >> 4) * 8 + r;
                float v = acc[tm][tn][r] + bv;
                if (RELU_BF16) {
                    v = v > 0.0f ? v : 0.0f;
                    outb[(size_t)m * Npad + n] = to_bf16(v);
                } else {
                    if (n < Nout) outf[(size_t)m * Nout + n] = v;
                }
            }
        }
    }
}

// ---------------------------------------------------------------------------
// Workspace layout (bytes, all 256-aligned): total ~ 93.2 MB
// ---------------------------------------------------------------------------
extern "C" void kernel_launch(void* const* d_in, const int* in_sizes, int n_in,
                              void* d_out, int out_size, void* d_ws, size_t ws_size,
                              hipStream_t stream) {
    const int*   x     = (const int*)d_in[0];
    const float* emb   = (const float*)d_in[1];
    const float* W1    = (const float*)d_in[2];
    const float* b1    = (const float*)d_in[3];
    const float* W2    = (const float*)d_in[4];
    const float* b2    = (const float*)d_in[5];
    const float* W_out = (const float*)d_in[6];
    const float* b_out = (const float*)d_in[7];
    float*       out   = (float*)d_out;

    char* ws = (char*)d_ws;
    size_t off = 0;
    bf16_t* feat = (bf16_t*)(ws + off); off += (size_t)kB  * kK1 * 2;
    bf16_t* h1   = (bf16_t*)(ws + off); off += (size_t)kB  * kH1 * 2;
    bf16_t* h2   = (bf16_t*)(ws + off); off += (size_t)kB  * kH2 * 2;
    bf16_t* W1b  = (bf16_t*)(ws + off); off += (size_t)kH1 * kK1 * 2;
    bf16_t* W2b  = (bf16_t*)(ws + off); off += (size_t)kH2 * kH1 * 2;
    bf16_t* W3b  = (bf16_t*)(ws + off); off += (size_t)kNC * kH2 * 2;

    // 1) weight conversion (re-done every call: deterministic, no cached state)
    {
        int tot1 = kH1 * kK1;
        convert_pad_kernel<<<(tot1 + 255) / 256, 256, 0, stream>>>(W1, W1b, kH1, 2 * kE, kH1, kK1);
        int tot2 = kH2 * kH1;
        convert_pad_kernel<<<(tot2 + 255) / 256, 256, 0, stream>>>(W2, W2b, kH2, kH1, kH2, kH1);
        int tot3 = kNC * kH2;
        convert_pad_kernel<<<(tot3 + 255) / 256, 256, 0, stream>>>(W_out, W3b, kC, kH2, kNC, kH2);
    }

    // 2) embedding + masked mean -> feat (bf16, K padded to 608)
    embed_feat_kernel<<<kB / 8, 256, 0, stream>>>(x, emb, feat);

    // 3) MLP via WMMA bf16 GEMMs (async-staged, double-buffered)
    dim3 g12(kH1 / 128, kB / 128);   // 16 x 64
    gemm_wmma_kernel<true><<<g12, 256, 0, stream>>>(feat, W1b, b1, h1, nullptr, kH1, kH1, kK1);
    gemm_wmma_kernel<true><<<g12, 256, 0, stream>>>(h1,   W2b, b2, h2, nullptr, kH2, kH2, kH1);
    dim3 g3(kNC / 128, kB / 128);    // 10 x 64
    gemm_wmma_kernel<false><<<g3, 256, 0, stream>>>(h2,   W3b, b_out, nullptr, out, kNC, kC, kH2);
}
